// NonLocalFilter_65489661329645
// MI455X (gfx1250) — matrix operand order repaired and checked by reading, not compiled, not run on hardware
//
#include <hip/hip_runtime.h>
#include <hip/hip_bf16.h>
#include <math.h>

typedef __attribute__((ext_vector_type(16))) __bf16 v16bf;
typedef __attribute__((ext_vector_type(8)))  __bf16 v8bf;
typedef __attribute__((ext_vector_type(8)))  float  v8f;

#define BB 16
#define CC 256
#define HH 48
#define NT (HH*HH)       // 2304
#define DD 128           // INTER
#define QTILE 64
#define KTILE 64
#define NBLK (NT/QTILE)  // 36

__device__ __forceinline__ v8f wmma_bf16(v16bf a, v16bf b, v8f c) {
  return __builtin_amdgcn_wmma_f32_16x16x32_bf16(false, a, false, b, (short)0, c, false, false);
}

// A-fragment (16x32 bf16): lane = row (lh); lo lanes hold K 0..15 contiguous, hi lanes K 16..31.
// p = pointer to this lane's row at the start of the 32-wide K chunk.
__device__ __forceinline__ v16bf load_afrag(const __bf16* p, int hiK) {
  v8bf lo = *(const v8bf*)(p + 2 * hiK);
  v8bf hi = *(const v8bf*)(p + 2 * hiK + 8);
  return __builtin_shufflevector(lo, hi, 0,1,2,3,4,5,6,7,8,9,10,11,12,13,14,15);
}

// B-fragment (32x16 bf16): lane = column; lo lanes hold K 0..15 contiguous, hi lanes K 16..31.
__device__ __forceinline__ v16bf load_bfrag(const __bf16* p, int hv) {
  return *(const v16bf*)(p + hv);
}

// ---------------------------------------------------------------------------
// Kernel 0: one-shot f32 -> bf16 conversion of the two weight matrices
// ---------------------------------------------------------------------------
__global__ void __launch_bounds__(256)
wcvt_kernel(const float* __restrict__ tw, const float* __restrict__ pw,
            __bf16* __restrict__ twb, __bf16* __restrict__ pwb)
{
  const int i = blockIdx.x * 256 + threadIdx.x;   // 0 .. 32767
  twb[i] = (__bf16)tw[i];
  pwb[i] = (__bf16)pw[i];
}

// ---------------------------------------------------------------------------
// Kernel 1: x -> bf16 copy (V matrix, [C,N] layout) + Q = scale*(theta@x + tb),
//           K = phi@x + pb, both stored row-major [N, D] in bf16.
// ---------------------------------------------------------------------------
__global__ void __launch_bounds__(128)
proj_kernel(const float* __restrict__ x, const __bf16* __restrict__ twb,
            const float* __restrict__ tb, const __bf16* __restrict__ pwb,
            const float* __restrict__ pb,
            __bf16* __restrict__ xbf, __bf16* __restrict__ qb, __bf16* __restrict__ kb)
{
  const int b    = blockIdx.y;
  const int n0   = blockIdx.x * QTILE;
  const int tid  = threadIdx.x;
  const int wave = tid >> 5;
  const int lane = tid & 31;
  const int lh   = lane & 15;
  const int hiK  = (lane & 16) ? 8 : 0;
  const int hv   = (lane & 16) ? 16 : 0;
  const int rbase= (lane & 16) ? 8 : 0;

  __shared__ __align__(16) __bf16 xt[32][QTILE + 8];  // [c-chunk][n-tile], padded

  const float* xb   = x   + (size_t)b * CC * NT;
  __bf16*      xbfb = xbf + (size_t)b * CC * NT;

  v8f vzero = {};
  v8f accq[8], acck[8];
  #pragma unroll
  for (int i = 0; i < 8; ++i) { accq[i] = vzero; acck[i] = vzero; }

  for (int cc = 0; cc < 8; ++cc) {
    const int c0 = cc * 32;
    __syncthreads();
    {
      // stage 32c x 64n tile: thread -> 16 consecutive n of one c row
      const int cl = tid >> 2;
      const int ns = (tid & 3) * 16;
      const float* src = xb + (size_t)(c0 + cl) * NT + n0 + ns;
      __align__(16) __bf16 tmp[16];
      #pragma unroll
      for (int i = 0; i < 16; ++i) tmp[i] = (__bf16)src[i];
      ((uint4*)&xt[cl][ns])[0] = ((const uint4*)tmp)[0];
      ((uint4*)&xt[cl][ns])[1] = ((const uint4*)tmp)[1];
      __bf16* gdst = xbfb + (size_t)(c0 + cl) * NT + n0 + ns;
      ((uint4*)gdst)[0] = ((const uint4*)tmp)[0];
      ((uint4*)gdst)[1] = ((const uint4*)tmp)[1];
    }
    __syncthreads();

    // A fragment: rows are n (x^T), K dim is c -> gather transposed from LDS
    const int row = wave * 16 + lh;
    v16bf af = {};
    #pragma unroll
    for (int j = 0; j < 8; ++j) {
      af[j]     = xt[2 * hiK + j][row];
      af[8 + j] = xt[2 * hiK + 8 + j][row];
    }

    #pragma unroll
    for (int kt = 0; kt < 8; ++kt) {
      // weights already bf16: single 32B B-fragment load per matrix
      v16bf bt = load_bfrag(twb + (size_t)(kt * 16 + lh) * CC + c0, hv);
      v16bf bp = load_bfrag(pwb + (size_t)(kt * 16 + lh) * CC + c0, hv);
      accq[kt] = wmma_bf16(af, bt, accq[kt]);
      acck[kt] = wmma_bf16(af, bp, acck[kt]);
    }
  }

  const float scale = 0.08838834764831845f;  // 1/sqrt(128), folded into Q
  #pragma unroll
  for (int kt = 0; kt < 8; ++kt) {
    const int kcol = kt * 16 + lh;
    const float bq = tb[kcol];
    const float bk = pb[kcol];
    #pragma unroll
    for (int r = 0; r < 8; ++r) {
      const size_t nrow = (size_t)b * NT + n0 + wave * 16 + rbase + r;
      qb[nrow * DD + kcol] = (__bf16)((accq[kt][r] + bq) * scale);
      kb[nrow * DD + kcol] = (__bf16)(acck[kt][r] + bk);
    }
  }
}

// ---------------------------------------------------------------------------
// Kernel 2: flash attention. out[b,c,n] = softmax_m(Q[n,:]·K[m,:]) · V[m,c]
// 1 wave = 16 query rows; block = 4 waves = 64 rows.
// ---------------------------------------------------------------------------
__global__ void __launch_bounds__(128)
flash_kernel(const __bf16* __restrict__ qb, const __bf16* __restrict__ kb,
             const __bf16* __restrict__ vb, float* __restrict__ out)
{
  const int b    = blockIdx.y;
  const int n0   = blockIdx.x * QTILE;
  const int tid  = threadIdx.x;
  const int wave = tid >> 5;
  const int lane = tid & 31;
  const int lh   = lane & 15;
  const int hiK  = (lane & 16) ? 8 : 0;
  const int hv   = (lane & 16) ? 16 : 0;
  const int rbase= (lane & 16) ? 8 : 0;

  __shared__ __align__(16) __bf16 pshare[4][16][KTILE + 8];  // per-wave P repack buffer

  // resident Q fragments (d=128 -> 4 chunks of 32)
  const __bf16* qrow = qb + (size_t)(b * NT + n0 + wave * 16 + lh) * DD;
  v16bf qf[4];
  #pragma unroll
  for (int ch = 0; ch < 4; ++ch) qf[ch] = load_afrag(qrow + ch * 32, hiK);

  v8f vzero = {};
  v8f o[16];                        // 16x256 f32 accumulator (per wave)
  #pragma unroll
  for (int i = 0; i < 16; ++i) o[i] = vzero;
  float m_s[8], l_s[8];
  #pragma unroll
  for (int r = 0; r < 8; ++r) { m_s[r] = -INFINITY; l_s[r] = 0.f; }

  const __bf16* kbase = kb + (size_t)b * NT * DD;
  const __bf16* vbase = vb + (size_t)b * CC * NT;

  for (int kt0 = 0; kt0 < NT; kt0 += KTILE) {
    // prefetch next key tile (K rows + V row segments) into cache while we compute
    if (kt0 + KTILE < NT) {
      const int nkt = kt0 + KTILE;
      const __bf16* pk = kbase + (size_t)(nkt + (wave & 1) * 32 + lane) * DD;
      __builtin_prefetch(pk, 0, 3);
      __builtin_prefetch(pk + 64, 0, 3);
      const __bf16* pv = vbase + (size_t)(wave * 64 + lane * 2) * NT + nkt;
      __builtin_prefetch(pv, 0, 3);
      __builtin_prefetch(pv + NT, 0, 3);
    }

    // ---- S = Q K^T for 16 rows x 64 keys ----
    v8f s[4];
    #pragma unroll
    for (int t = 0; t < 4; ++t) s[t] = vzero;
    #pragma unroll
    for (int t = 0; t < 4; ++t) {
      const __bf16* krow = kbase + (size_t)(kt0 + t * 16 + lh) * DD;
      #pragma unroll
      for (int ch = 0; ch < 4; ++ch) {
        v16bf kf = load_bfrag(krow + ch * 32, hv);
        s[t] = wmma_bf16(qf[ch], kf, s[t]);
      }
    }

    // ---- online softmax (rows live across 16-lane halves) ----
    float mrow[8];
    #pragma unroll
    for (int r = 0; r < 8; ++r)
      mrow[r] = fmaxf(fmaxf(s[0][r], s[1][r]), fmaxf(s[2][r], s[3][r]));
    #pragma unroll
    for (int off = 1; off < 16; off <<= 1) {
      #pragma unroll
      for (int r = 0; r < 8; ++r)
        mrow[r] = fmaxf(mrow[r], __shfl_xor(mrow[r], off, 32));
    }
    float alpha[8];
    #pragma unroll
    for (int r = 0; r < 8; ++r) {
      const float mn = fmaxf(m_s[r], mrow[r]);
      alpha[r] = __expf(m_s[r] - mn);
      m_s[r] = mn;
    }
    float rs[8];
    #pragma unroll
    for (int r = 0; r < 8; ++r) rs[r] = 0.f;
    #pragma unroll
    for (int t = 0; t < 4; ++t) {
      #pragma unroll
      for (int r = 0; r < 8; ++r) {
        const float p = __expf(s[t][r] - m_s[r]);
        s[t][r] = p;
        rs[r] += p;
      }
    }
    #pragma unroll
    for (int off = 1; off < 16; off <<= 1) {
      #pragma unroll
      for (int r = 0; r < 8; ++r)
        rs[r] += __shfl_xor(rs[r], off, 32);
    }
    #pragma unroll
    for (int r = 0; r < 8; ++r) l_s[r] = l_s[r] * alpha[r] + rs[r];
    #pragma unroll
    for (int ct = 0; ct < 16; ++ct) {
      #pragma unroll
      for (int r = 0; r < 8; ++r) o[ct][r] *= alpha[r];
    }

    // ---- repack P (C/D layout -> A layout) via per-wave LDS buffer ----
    #pragma unroll
    for (int t = 0; t < 4; ++t) {
      #pragma unroll
      for (int r = 0; r < 8; ++r)
        pshare[wave][rbase + r][t * 16 + lh] = (__bf16)s[t][r];
    }
    // ---- O += P V ----
    #pragma unroll
    for (int mch = 0; mch < 2; ++mch) {
      const __bf16* prow = &pshare[wave][lh][mch * 32];
      v16bf pf = load_afrag(prow, hiK);
      #pragma unroll
      for (int ct = 0; ct < 16; ++ct) {
        const __bf16* vrow = vbase + (size_t)(ct * 16 + lh) * NT + kt0 + mch * 32;
        v16bf vf = load_bfrag(vrow, hv);
        o[ct] = wmma_bf16(pf, vf, o[ct]);
      }
    }
  }

  // ---- normalize and store: out[b][c][n], 8 consecutive n per lane per tile ----
  float inv[8];
  #pragma unroll
  for (int r = 0; r < 8; ++r) inv[r] = 1.f / l_s[r];
  float* ob = out + (size_t)b * CC * NT;
  #pragma unroll
  for (int ct = 0; ct < 16; ++ct) {
    v8f res;
    #pragma unroll
    for (int r = 0; r < 8; ++r) res[r] = o[ct][r] * inv[r];
    *(v8f*)(ob + (size_t)(ct * 16 + lh) * NT + n0 + wave * 16 + rbase) = res;
  }
}

extern "C" void kernel_launch(void* const* d_in, const int* in_sizes, int n_in,
                              void* d_out, int out_size, void* d_ws, size_t ws_size,
                              hipStream_t stream) {
  (void)in_sizes; (void)n_in; (void)out_size; (void)ws_size;
  const float* x  = (const float*)d_in[0];
  const float* tw = (const float*)d_in[1];
  const float* tb = (const float*)d_in[2];
  const float* pw = (const float*)d_in[3];
  const float* pb = (const float*)d_in[4];
  float* out = (float*)d_out;

  __bf16* xbf = (__bf16*)d_ws;                       // [B][C][N]  bf16 (V)
  __bf16* qb  = xbf + (size_t)BB * CC * NT;          // [B][N][D]  bf16
  __bf16* kb  = qb  + (size_t)BB * NT * DD;          // [B][N][D]  bf16
  __bf16* twb = kb  + (size_t)BB * NT * DD;          // [D][C]     bf16
  __bf16* pwb = twb + (size_t)DD * CC;               // [D][C]     bf16

  wcvt_kernel<<<(DD * CC) / 256, 256, 0, stream>>>(tw, pw, twb, pwb);
  dim3 grid(NBLK, BB);
  proj_kernel<<<grid, 128, 0, stream>>>(x, twb, tb, pwb, pb, xbf, qb, kb);
  flash_kernel<<<grid, 128, 0, stream>>>(qb, kb, xbf, out);
}
